// LateralFusion_5111011082603
// MI455X (gfx1250) — matrix operand order repaired and checked by reading, not compiled, not run on hardware
//
// MI455X (gfx1250, CDNA5) implementation of LateralFusion forward.
// All large GEMMs via v_wmma_f32_16x16x32_bf16 (bf16 A/B, fp32 accum).
// wave32, 256-thread workgroups (8 waves), 128x128x32 tiles staged in LDS.
// Weights are transposed once per launch so BOTH GEMM operands stage with
// b128 transfers (no LDS scatter). Staging uses CDNA5 async global->LDS DMA
// (global_load_async_to_lds_b128 + s_wait_asynccnt) when available.

#include <hip/hip_runtime.h>
#include <math.h>

#define DIMV   1024
#define NHV    8
#define HDV    128
#define KPTS   4
#define SSV    32
#define S_TOK  1024
#define BATCH  16
#define NTOK   (BATCH * S_TOK)   // 16384
#define FFNV   (4 * DIMV)        // 4096

typedef __bf16 bf16;
typedef __attribute__((ext_vector_type(16))) __bf16 v16bf;
typedef __attribute__((ext_vector_type(8)))  float   v8f;

union Frag16 { v16bf v; uint4 u[2]; };

#if defined(__has_builtin)
#if __has_builtin(__builtin_amdgcn_global_load_async_to_lds_b128) && \
    __has_builtin(__builtin_amdgcn_s_wait_asynccnt)
#define USE_ASYNC_LDS 1
#endif
#endif

#ifdef USE_ASYNC_LDS
typedef int v4i __attribute__((ext_vector_type(4)));
typedef __attribute__((address_space(1))) v4i glb_v4i;  // global int4*
typedef __attribute__((address_space(3))) v4i lds_v4i;  // LDS int4*
#endif

// ---------------------------------------------------------------- GEMM ----
// C(MxN) = epilogue( A(MxK)bf16 @ Bt(NxK)bf16 + bias + Res )
// Bt is the transposed weight (N x K, row-major == original B column-major).
// EPI: 0=none 1=gelu(exact) 2=tanh ; OUT_BF: 0=f32 out, 1=bf16 out
#define BM 128
#define BN 128
#define BKK 32
#define LDK 40   // padded LDS K-stride (elements); 80B rows keep 16B alignment

template <int EPI, int OUT_BF>
__global__ __launch_bounds__(256) void gemm_k(
    const bf16* __restrict__ A, const bf16* __restrict__ Bt,
    const float* __restrict__ bias, const float* __restrict__ Res,
    void* __restrict__ Cout, int M, int N, int K)
{
  __shared__ bf16 lA[BM * LDK];
  __shared__ bf16 lB[BN * LDK];

  const int tid   = threadIdx.x;
  const int lane  = tid & 31;
  const int wid   = tid >> 5;
  const int lmod  = lane & 15;
  const int lhalf = lane >> 4;
  const int wm    = wid & 3;   // 4 waves along M (32 rows each)
  const int wn    = wid >> 2;  // 2 waves along N (64 cols each)
  const int m0    = blockIdx.y * BM;
  const int n0    = blockIdx.x * BN;

  v8f acc[2][4];
#pragma unroll
  for (int i = 0; i < 2; i++)
#pragma unroll
    for (int j = 0; j < 4; j++)
#pragma unroll
      for (int e = 0; e < 8; e++) acc[i][j][e] = 0.0f;

  // Both tiles are 128 rows x 32 K-elems: thread -> (row = tid>>1, 16-elem chunk)
  const int srow = tid >> 1, sch = tid & 1;
  const bf16* ga  = A  + (size_t)(m0 + srow) * K + sch * 16;
  const bf16* gbt = Bt + (size_t)(n0 + srow) * K + sch * 16;
  bf16* la = &lA[srow * LDK + sch * 16];
  bf16* lb = &lB[srow * LDK + sch * 16];

  for (int k0 = 0; k0 < K; k0 += BKK) {
#ifdef USE_ASYNC_LDS
    // CDNA5 async DMA: global -> LDS without touching VGPRs (ASYNCcnt).
    __builtin_amdgcn_global_load_async_to_lds_b128(
        (glb_v4i*)(ga + k0),      (lds_v4i*)la,       0, 0);
    __builtin_amdgcn_global_load_async_to_lds_b128(
        (glb_v4i*)(ga + k0 + 8),  (lds_v4i*)(la + 8), 0, 0);
    __builtin_amdgcn_global_load_async_to_lds_b128(
        (glb_v4i*)(gbt + k0),     (lds_v4i*)lb,       0, 0);
    __builtin_amdgcn_global_load_async_to_lds_b128(
        (glb_v4i*)(gbt + k0 + 8), (lds_v4i*)(lb + 8), 0, 0);
    __builtin_amdgcn_s_wait_asynccnt(0);
#else
    {
      uint4 a0 = *(const uint4*)(ga + k0);
      uint4 a1 = *(const uint4*)(ga + k0 + 8);
      uint4 b0 = *(const uint4*)(gbt + k0);
      uint4 b1 = *(const uint4*)(gbt + k0 + 8);
      *(uint4*)la       = a0;
      *(uint4*)(la + 8) = a1;
      *(uint4*)lb       = b0;
      *(uint4*)(lb + 8) = b1;
    }
#endif
    if (k0 + BKK < K) {                     // global_prefetch next k-tiles
      __builtin_prefetch(ga + k0 + BKK, 0, 0);
      __builtin_prefetch(gbt + k0 + BKK, 0, 0);
    }
    __syncthreads();

    // Preload all fragments, then issue the WMMA burst back-to-back.
    Frag16 fa[2], fb[4];
#pragma unroll
    for (int im = 0; im < 2; im++) {
      // 16-bit A layout: lanes<16 hold K {0..7,16..23}, lanes>=16 {8..15,24..31}
      const bf16* pa = &lA[(wm * 32 + im * 16 + lmod) * LDK + lhalf * 8];
      fa[im].u[0] = *(const uint4*)pa;
      fa[im].u[1] = *(const uint4*)(pa + 16);
    }
#pragma unroll
    for (int in = 0; in < 4; in++) {
      // 16-bit B layout: lane holds col lane%16, lanes<16 K {0..7,16..23}, lanes>=16 {8..15,24..31}
      const bf16* pb = &lB[(wn * 64 + in * 16 + lmod) * LDK + lhalf * 8];
      fb[in].u[0] = *(const uint4*)pb;
      fb[in].u[1] = *(const uint4*)(pb + 16);
    }
#pragma unroll
    for (int in = 0; in < 4; in++)
#pragma unroll
      for (int im = 0; im < 2; im++)
        acc[im][in] = __builtin_amdgcn_wmma_f32_16x16x32_bf16(
            false, fa[im].v, false, fb[in].v, (short)0, acc[im][in], false, false);
    __syncthreads();
  }

  float* Cf = (float*)Cout;
  bf16*  Cb = (bf16*)Cout;
#pragma unroll
  for (int im = 0; im < 2; im++) {
#pragma unroll
    for (int in = 0; in < 4; in++) {
      const int col   = n0 + wn * 64 + in * 16 + lmod;
      const int rbase = m0 + wm * 32 + im * 16 + lhalf * 8;
      const float bsv = bias ? bias[col] : 0.0f;
#pragma unroll
      for (int i = 0; i < 8; i++) {
        const size_t idx = (size_t)(rbase + i) * N + col;
        float v = acc[im][in][i] + bsv;
        if (Res) v += Res[idx];
        if (EPI == 1) v = 0.5f * v * (1.0f + erff(v * 0.70710678118654752f));
        else if (EPI == 2) v = tanhf(v);
        if (OUT_BF) Cb[idx] = (bf16)v; else Cf[idx] = v;
      }
    }
  }
}

// ------------------------------------------------------------ LayerNorm ----
__global__ __launch_bounds__(256) void layernorm_k(
    const float* __restrict__ X, const float* __restrict__ g,
    const float* __restrict__ bta, bf16* __restrict__ Yb, float* __restrict__ Yf)
{
  __shared__ float s1[256], s2[256];
  const int row = blockIdx.x;
  const float* x = X + (size_t)row * DIMV;
  float v[4], a = 0.0f, b2 = 0.0f;
#pragma unroll
  for (int i = 0; i < 4; i++) {
    v[i] = x[threadIdx.x + 256 * i];
    a += v[i]; b2 += v[i] * v[i];
  }
  s1[threadIdx.x] = a; s2[threadIdx.x] = b2;
  __syncthreads();
  for (int off = 128; off > 0; off >>= 1) {
    if (threadIdx.x < off) {
      s1[threadIdx.x] += s1[threadIdx.x + off];
      s2[threadIdx.x] += s2[threadIdx.x + off];
    }
    __syncthreads();
  }
  const float mean = s1[0] / DIMV;
  const float var  = s2[0] / DIMV - mean * mean;
  const float rstd = rsqrtf(var + 1e-5f);
#pragma unroll
  for (int i = 0; i < 4; i++) {
    const int c = threadIdx.x + 256 * i;
    const float y = (v[i] - mean) * rstd * g[c] + bta[c];
    if (Yb) Yb[(size_t)row * DIMV + c] = (bf16)y;
    if (Yf) Yf[(size_t)row * DIMV + c] = y;
  }
}

// ------------------------------------------------------------ elementwise --
__global__ __launch_bounds__(256) void add_pos_k(
    const float* __restrict__ x, const float* __restrict__ pos,
    float* __restrict__ y, int n, int pmask)
{
  const int i = blockIdx.x * 256 + threadIdx.x;
  if (i < n) y[i] = x[i] + pos[i & pmask];
}

// Tiled fp32 -> bf16 transpose: y[n*K + k] = bf16(x[k*N + n]).
// Coalesced reads and writes via a 32x33 LDS tile; runs once per weight.
__global__ __launch_bounds__(256) void cvt_transpose_k(
    const float* __restrict__ x, bf16* __restrict__ y, int K, int N)
{
  __shared__ float t[32][33];
  const int k0 = blockIdx.y * 32, n0 = blockIdx.x * 32;
  const int tx = threadIdx.x & 31, ty = threadIdx.x >> 5;  // 32 x 8
#pragma unroll
  for (int i = 0; i < 32; i += 8)
    t[ty + i][tx] = x[(size_t)(k0 + ty + i) * N + n0 + tx];
  __syncthreads();
#pragma unroll
  for (int i = 0; i < 32; i += 8)
    y[(size_t)(n0 + ty + i) * K + k0 + tx] = (bf16)t[tx][ty + i];
}

// --------------------------------------------- deformable sampling core ----
// One 128-thread block per (b, s, head): offset MLP + softmax weights +
// bilinear gather from v, producing head_out (bf16) for the Wout GEMM.
__global__ __launch_bounds__(128) void deform_k(
    const float* __restrict__ Q, const float* __restrict__ V,
    const float* __restrict__ Wo1, const float* __restrict__ bo1,
    const float* __restrict__ Wo2, const float* __restrict__ bo2,
    const float* __restrict__ Ww,  const float* __restrict__ bw,
    bf16* __restrict__ HO)
{
  __shared__ float qs[HDV], h1s[64], offs[8], wk[4];
  const int g = blockIdx.x;
  const int h = g & (NHV - 1);
  const int s = (g >> 3) & (S_TOK - 1);
  const int b = g >> 13;
  const int tid = threadIdx.x;

  qs[tid] = Q[((size_t)(b * S_TOK + s)) * DIMV + h * HDV + tid];
  __syncthreads();

  if (tid < 64) {
    float a = bo1[tid];
    for (int i = 0; i < HDV; i++) a += qs[i] * Wo1[i * 64 + tid];
    h1s[tid] = fmaxf(a, 0.0f);
  }
  __syncthreads();

  if (tid < 8) {
    float a = bo2[tid];
    for (int e = 0; e < 64; e++) a += h1s[e] * Wo2[e * 8 + tid];
    offs[tid] = tanhf(a) * 0.5f;
  } else if (tid < 12) {
    const int k = tid - 8;
    float a = bw[k];
    for (int i = 0; i < HDV; i++) a += qs[i] * Ww[i * KPTS + k];
    wk[k] = a;
  }
  __syncthreads();

  if (tid == 0) {
    const float m = fmaxf(fmaxf(wk[0], wk[1]), fmaxf(wk[2], wk[3]));
    float e0 = expf(wk[0] - m), e1 = expf(wk[1] - m);
    float e2 = expf(wk[2] - m), e3 = expf(wk[3] - m);
    const float inv = 1.0f / (e0 + e1 + e2 + e3);
    wk[0] = e0 * inv; wk[1] = e1 * inv; wk[2] = e2 * inv; wk[3] = e3 * inv;
  }
  __syncthreads();

  const float gx = -1.0f + 2.0f * (float)(s & 31) / 31.0f;
  const float gy = -1.0f + 2.0f * (float)(s >> 5) / 31.0f;
  const float* vb = V + (size_t)b * S_TOK * DIMV + h * HDV + tid;
  float acc = 0.0f;
#pragma unroll
  for (int k = 0; k < KPTS; k++) {
    const float ix = (gx + offs[2 * k]     + 1.0f) * 0.5f * 31.0f;
    const float iy = (gy + offs[2 * k + 1] + 1.0f) * 0.5f * 31.0f;
    const float x0 = floorf(ix), y0 = floorf(iy);
#pragma unroll
    for (int dy = 0; dy < 2; dy++) {
#pragma unroll
      for (int dx = 0; dx < 2; dx++) {
        const float xc = x0 + dx, yc = y0 + dy;
        const float w = (1.0f - fabsf(ix - xc)) * (1.0f - fabsf(iy - yc));
        if (xc >= 0.0f && xc <= 31.0f && yc >= 0.0f && yc <= 31.0f) {
          const int idx = (int)yc * SSV + (int)xc;
          acc += wk[k] * w * vb[(size_t)idx * DIMV];
        }
      }
    }
  }
  HO[((size_t)(b * S_TOK + s)) * DIMV + h * HDV + tid] = (bf16)acc;
}

// --------------------------------------------------------------- pooling ---
__global__ __launch_bounds__(256) void pool_score_k(
    const bf16* __restrict__ S1, const float* __restrict__ Wp2,
    const float* __restrict__ bp2, float* __restrict__ score)
{
  const int t = blockIdx.x * 256 + threadIdx.x;
  float a = bp2[0];
  const bf16* row = S1 + (size_t)t * (DIMV / 4);
  for (int e = 0; e < DIMV / 4; e++) a += (float)row[e] * Wp2[e];
  score[t] = a;
}

__global__ __launch_bounds__(256) void pool_softmax_k(
    const float* __restrict__ score, float* __restrict__ probs)
{
  __shared__ float sh[256];
  const int b = blockIdx.x;
  const float* sc = score + b * S_TOK;
  float m = -1e30f;
  for (int i = threadIdx.x; i < S_TOK; i += 256) m = fmaxf(m, sc[i]);
  sh[threadIdx.x] = m; __syncthreads();
  for (int off = 128; off > 0; off >>= 1) {
    if (threadIdx.x < off) sh[threadIdx.x] = fmaxf(sh[threadIdx.x], sh[threadIdx.x + off]);
    __syncthreads();
  }
  m = sh[0]; __syncthreads();
  float sum = 0.0f;
  for (int i = threadIdx.x; i < S_TOK; i += 256) sum += expf(sc[i] - m);
  sh[threadIdx.x] = sum; __syncthreads();
  for (int off = 128; off > 0; off >>= 1) {
    if (threadIdx.x < off) sh[threadIdx.x] += sh[threadIdx.x + off];
    __syncthreads();
  }
  const float inv = 1.0f / sh[0];
  for (int i = threadIdx.x; i < S_TOK; i += 256)
    probs[b * S_TOK + i] = expf(sc[i] - m) * inv;
}

__global__ __launch_bounds__(256) void pool_wsum_k(
    const float* __restrict__ probs, const float* __restrict__ X,
    float* __restrict__ out)
{
  const int gid = blockIdx.x * 256 + threadIdx.x;  // b*DIM + d
  const int b = gid >> 10, d = gid & 1023;
  const float* x = X + (size_t)b * S_TOK * DIMV + d;
  const float* p = probs + b * S_TOK;
  float a = 0.0f;
  for (int s = 0; s < S_TOK; s++) a += p[s] * x[(size_t)s * DIMV];
  out[gid] = a;
}

__global__ __launch_bounds__(256) void fuse_k(
    const float* __restrict__ ccp, const float* __restrict__ mlop,
    const float* __restrict__ Wfu, const float* __restrict__ bfu,
    float* __restrict__ fused)
{
  const int gid = blockIdx.x * 256 + threadIdx.x;  // b*DIM + n
  const int b = gid >> 10, n = gid & 1023;
  float a = bfu[n];
  const float* c = ccp + b * DIMV;
  const float* m = mlop + b * DIMV;
  for (int i = 0; i < DIMV; i++) a += c[i] * Wfu[(size_t)i * DIMV + n];
  for (int i = 0; i < DIMV; i++) a += m[i] * Wfu[(size_t)(DIMV + i) * DIMV + n];
  fused[gid] = a;
}

__global__ __launch_bounds__(256) void lngelu_k(
    const float* __restrict__ X, const float* __restrict__ g,
    const float* __restrict__ bta, float* __restrict__ out)
{
  __shared__ float s1[256], s2[256];
  const int row = blockIdx.x;
  const float* x = X + (size_t)row * DIMV;
  float v[4], a = 0.0f, b2 = 0.0f;
#pragma unroll
  for (int i = 0; i < 4; i++) {
    v[i] = x[threadIdx.x + 256 * i];
    a += v[i]; b2 += v[i] * v[i];
  }
  s1[threadIdx.x] = a; s2[threadIdx.x] = b2;
  __syncthreads();
  for (int off = 128; off > 0; off >>= 1) {
    if (threadIdx.x < off) {
      s1[threadIdx.x] += s1[threadIdx.x + off];
      s2[threadIdx.x] += s2[threadIdx.x + off];
    }
    __syncthreads();
  }
  const float mean = s1[0] / DIMV;
  const float var  = s2[0] / DIMV - mean * mean;
  const float rstd = rsqrtf(var + 1e-5f);
#pragma unroll
  for (int i = 0; i < 4; i++) {
    const int c = threadIdx.x + 256 * i;
    float y = (v[i] - mean) * rstd * g[c] + bta[c];
    y = 0.5f * y * (1.0f + erff(y * 0.70710678118654752f));
    out[(size_t)row * DIMV + c] = y;
  }
}

// ------------------------------------------------------------ host side ----
extern "C" void kernel_launch(void* const* d_in, const int* in_sizes, int n_in,
                              void* d_out, int out_size, void* d_ws, size_t ws_size,
                              hipStream_t stream)
{
  (void)in_sizes; (void)n_in; (void)out_size; (void)ws_size;

  // Input layout (setup_inputs dict order, params flattened in insertion order):
  // 0 cc_feat, 1 mlo_feat, 2 pos_embed, 3.. blocks[0..3] (22 each),
  // 91 fn_g, 92 fn_b, 93 Wp1, 94 bp1, 95 Wp2, 96 bp2, 97 Wfu, 98 bfu,
  // 99 lnfu_g, 100 lnfu_b
  const float* cc_feat  = (const float*)d_in[0];
  const float* mlo_feat = (const float*)d_in[1];
  const float* pos      = (const float*)d_in[2];
  auto BP = [&](int blk, int j) -> const float* {
    return (const float*)d_in[3 + 22 * blk + j];
  };
  const int O = 3 + 22 * 4;
  const float* fn_g   = (const float*)d_in[O + 0];
  const float* fn_b   = (const float*)d_in[O + 1];
  const float* Wp1    = (const float*)d_in[O + 2];
  const float* bp1    = (const float*)d_in[O + 3];
  const float* Wp2    = (const float*)d_in[O + 4];
  const float* bp2    = (const float*)d_in[O + 5];
  const float* Wfu    = (const float*)d_in[O + 6];
  const float* bfu    = (const float*)d_in[O + 7];
  const float* lnfu_g = (const float*)d_in[O + 8];
  const float* lnfu_b = (const float*)d_in[O + 9];

  // --- workspace bump allocator (256B aligned) ---
  char* wptr = (char*)d_ws;
  auto alloc = [&](size_t bytes) -> void* {
    void* p = wptr;
    wptr += (bytes + 255) & ~(size_t)255;
    return p;
  };
  const size_t AF = (size_t)NTOK * DIMV * sizeof(float);   // 64 MB
  const size_t AB = (size_t)NTOK * DIMV * sizeof(bf16);    // 32 MB
  float* act[4];
  for (int i = 0; i < 4; i++) act[i] = (float*)alloc(AF);
  float* qbuf = (float*)alloc(AF);
  float* vbuf = (float*)alloc(AF);
  bf16*  s0   = (bf16*)alloc(AB);
  bf16*  s1   = (bf16*)alloc(AB);
  bf16*  hbuf = (bf16*)alloc((size_t)NTOK * FFNV * sizeof(bf16));  // 128 MB
  bf16*  wq_bf[4], *wv_bf[4], *wo_bf[4], *wf1_bf[4], *wf2_bf[4];
  for (int b = 0; b < 4; b++) {
    wq_bf[b]  = (bf16*)alloc((size_t)DIMV * DIMV * sizeof(bf16));
    wv_bf[b]  = (bf16*)alloc((size_t)DIMV * DIMV * sizeof(bf16));
    wo_bf[b]  = (bf16*)alloc((size_t)DIMV * DIMV * sizeof(bf16));
    wf1_bf[b] = (bf16*)alloc((size_t)DIMV * FFNV * sizeof(bf16));
    wf2_bf[b] = (bf16*)alloc((size_t)FFNV * DIMV * sizeof(bf16));
  }
  bf16*  wp1_bf = (bf16*)alloc((size_t)DIMV * (DIMV / 4) * sizeof(bf16));
  bf16*  s1hid  = (bf16*)alloc((size_t)NTOK * (DIMV / 4) * sizeof(bf16));
  float* score  = (float*)alloc(NTOK * sizeof(float));
  float* probs  = (float*)alloc(NTOK * sizeof(float));
  float* ccp    = (float*)alloc(BATCH * DIMV * sizeof(float));
  float* mlop   = (float*)alloc(BATCH * DIMV * sizeof(float));
  float* fusedb = (float*)alloc(BATCH * DIMV * sizeof(float));

  // Transpose+convert weights once per launch: W (KxN) -> Wt (NxK) bf16.
  auto cvtT = [&](const float* src, bf16* dst, int K, int N) {
    cvt_transpose_k<<<dim3(N / 32, K / 32), 256, 0, stream>>>(src, dst, K, N);
  };
  for (int b = 0; b < 4; b++) {
    cvtT(BP(b, 6),  wq_bf[b],  DIMV, DIMV);
    cvtT(BP(b, 8),  wv_bf[b],  DIMV, DIMV);
    cvtT(BP(b, 10), wo_bf[b],  DIMV, DIMV);
    cvtT(BP(b, 18), wf1_bf[b], DIMV, FFNV);
    cvtT(BP(b, 20), wf2_bf[b], FFNV, DIMV);
  }
  cvtT(Wp1, wp1_bf, DIMV, DIMV / 4);

  // --- add positional embedding ---
  const int nelem = NTOK * DIMV;
  add_pos_k<<<nelem / 256, 256, 0, stream>>>(cc_feat,  pos, act[0], nelem, S_TOK * DIMV - 1);
  add_pos_k<<<nelem / 256, 256, 0, stream>>>(mlo_feat, pos, act[1], nelem, S_TOK * DIMV - 1);

  float* cc = act[0]; float* mlo = act[1];
  float* ncc = act[2]; float* nmlo = act[3];

  auto gemm_grid = [](int M, int N) { return dim3(N / BN, M / BM); };

  for (int blk = 0; blk < 4; blk++) {
    const float* src = (blk & 1) ? mlo : cc;
    const float* tgt = (blk & 1) ? cc  : mlo;
    float*       dst = (blk & 1) ? nmlo : ncc;

    // LayerNorms -> bf16 GEMM inputs
    layernorm_k<<<NTOK, 256, 0, stream>>>(src, BP(blk, 0), BP(blk, 1), s0, nullptr);
    layernorm_k<<<NTOK, 256, 0, stream>>>(tgt, BP(blk, 2), BP(blk, 3), s1, nullptr);
    // q = qn@Wq + bq ; v = vn@Wv + bv   (fp32 out: feeds sampling)
    gemm_k<0, 0><<<gemm_grid(NTOK, DIMV), 256, 0, stream>>>(
        s0, wq_bf[blk], BP(blk, 7), nullptr, qbuf, NTOK, DIMV, DIMV);
    gemm_k<0, 0><<<gemm_grid(NTOK, DIMV), 256, 0, stream>>>(
        s1, wv_bf[blk], BP(blk, 9), nullptr, vbuf, NTOK, DIMV, DIMV);
    // deformable sampling -> head_out (bf16)
    deform_k<<<BATCH * S_TOK * NHV, 128, 0, stream>>>(
        qbuf, vbuf, BP(blk, 12), BP(blk, 13), BP(blk, 14), BP(blk, 15),
        BP(blk, 16), BP(blk, 17), s0);
    // out = src + head_out@Wout + bout
    gemm_k<0, 0><<<gemm_grid(NTOK, DIMV), 256, 0, stream>>>(
        s0, wo_bf[blk], BP(blk, 11), src, dst, NTOK, DIMV, DIMV);
    // FFN: hn = LN(out); h = gelu(hn@Wf1+bf1); out += h@Wf2+bf2
    layernorm_k<<<NTOK, 256, 0, stream>>>(dst, BP(blk, 4), BP(blk, 5), s1, nullptr);
    gemm_k<1, 1><<<gemm_grid(NTOK, FFNV), 256, 0, stream>>>(
        s1, wf1_bf[blk], BP(blk, 19), nullptr, hbuf, NTOK, FFNV, DIMV);
    gemm_k<0, 0><<<gemm_grid(NTOK, DIMV), 256, 0, stream>>>(
        hbuf, wf2_bf[blk], BP(blk, 21), dst, dst, NTOK, DIMV, FFNV);

    if (blk & 1) {  // both directions of this layer done -> swap
      float* t;
      t = cc;  cc  = ncc;  ncc  = t;
      t = mlo; mlo = nmlo; nmlo = t;
    }
  }

  // --- final norm + attention pooling for each stream ---
  auto pool = [&](const float* x, float* out_p) {
    layernorm_k<<<NTOK, 256, 0, stream>>>(x, fn_g, fn_b, s0, qbuf);
    gemm_k<2, 1><<<gemm_grid(NTOK, DIMV / 4), 256, 0, stream>>>(
        s0, wp1_bf, bp1, nullptr, s1hid, NTOK, DIMV / 4, DIMV);
    pool_score_k<<<NTOK / 256, 256, 0, stream>>>(s1hid, Wp2, bp2, score);
    pool_softmax_k<<<BATCH, 256, 0, stream>>>(score, probs);
    pool_wsum_k<<<(BATCH * DIMV) / 256, 256, 0, stream>>>(probs, qbuf, out_p);
  };
  pool(cc, ccp);
  pool(mlo, mlop);

  // --- fuse + LN + GELU -> d_out ---
  fuse_k<<<(BATCH * DIMV) / 256, 256, 0, stream>>>(ccp, mlop, Wfu, bfu, fusedb);
  lngelu_k<<<BATCH, 256, 0, stream>>>(fusedb, lnfu_g, lnfu_b, (float*)d_out);
}
